// MultiHeadAttention_18202071400711
// MI455X (gfx1250) — matrix operand order repaired and checked by reading, not compiled, not run on hardware
//
#include <hip/hip_runtime.h>
#include <hip/hip_bf16.h>

constexpr int Hn = 4;
constexpr int Dm = 256;
constexpr int HD = 64;
constexpr int Bn = 2;
constexpr int Sn = 4096;
constexpr int BS = Bn * Sn;

typedef __attribute__((ext_vector_type(16))) _Float16 v16h;
typedef __attribute__((ext_vector_type(8)))  float    v8f;

// ---- CDNA5 async global->LDS copy (ASYNCcnt path) ------------------------
// VDST = VGPR with LDS byte address (low 32 bits of generic shared pointer),
// VADDR = 64-bit global address, SADDR = off (GV mode).
__device__ __forceinline__ void async_b128(void* lds_dst, const void* gsrc) {
  asm volatile("global_load_async_to_lds_b128 %0, %1, off"
               :: "v"((unsigned)(uintptr_t)lds_dst), "v"(gsrc)
               : "memory");
}
__device__ __forceinline__ void wait_async0() {
  asm volatile("s_wait_asynccnt 0" ::: "memory");
}
// compiler-only reordering fence (same-wave LDS RAW is in-order in HW)
__device__ __forceinline__ void compiler_fence() {
  asm volatile("" ::: "memory");
}

// ---- WMMA fragment loaders (layouts per CDNA5 ISA 7.12.2) ----------------

// A-matrix 16x32 f16: lane m=lane&15, half=lane>>4;
//   a[0..7]  = A[m][kbase + half*8 + i] ; a[8..15] = A[m][kbase + 16 + half*8 + i]
__device__ __forceinline__ v16h load_a_h(const _Float16* base, int ld, int kbase) {
  const int lane = threadIdx.x & 31;
  const int m = lane & 15, half = lane >> 4;
  const _Float16* p0 = base + (size_t)m * ld + kbase + half * 8;
  const _Float16* p1 = p0 + 16;
  v16h a;
#pragma unroll
  for (int i = 0; i < 8; ++i) { a[i] = p0[i]; a[8 + i] = p1[i]; }
  return a;
}

// B-matrix 32x16 f16: lane n=lane&15 holds column n0+n; b[0..7] = B[kbase+half*8+i][n0+n] ; b[8..15] = B[kbase+16+half*8+i][n0+n]
// (source is row-major [N][K], so per-lane two runs of 8 contiguous halfs, 16 apart).
__device__ __forceinline__ v16h load_b_h(const _Float16* base, int ld, int n0, int kbase) {
  const int lane = threadIdx.x & 31;
  const int n = lane & 15, half = lane >> 4;
  const _Float16* p = base + (size_t)(n0 + n) * ld + kbase + half * 8;
  v16h b;
#pragma unroll
  for (int i = 0; i < 16; ++i) b[i] = p[(i < 8) ? i : i + 8];
  return b;
}

__device__ __forceinline__ v8f wmma16(v16h a, v16h b, v8f c) {
  return __builtin_amdgcn_wmma_f32_16x16x32_f16(false, a, false, b, (short)0, c, false, false);
}

// cooperative: stage a 16x256 fp32 row-tile as f16 into padded LDS tile
// (256 threads, each converts 16 consecutive elements of one row)
__device__ __forceinline__ void stage_a_tile_f16(_Float16* dst264, const float* src, int ld) {
  const int t = threadIdx.x;
  const int r = t >> 4, c0 = (t & 15) * 16;
  float tmp[16];
#pragma unroll
  for (int i = 0; i < 16; ++i) tmp[i] = src[(size_t)r * ld + c0 + i];
#pragma unroll
  for (int i = 0; i < 16; ++i) dst264[r * 264 + c0 + i] = (_Float16)tmp[i];
}

// ---- Kernel 0: weights fp32 -> f16 ---------------------------------------
__global__ void cvt_weights_kernel(const float* __restrict__ wq, const float* __restrict__ wk,
                                   const float* __restrict__ wv, const float* __restrict__ wo,
                                   _Float16* __restrict__ w16) {
  int i = blockIdx.x * blockDim.x + threadIdx.x;
  if (i < Dm * Dm) {
    w16[i]               = (_Float16)wq[i];
    w16[Dm * Dm + i]     = (_Float16)wk[i];
    w16[2 * Dm * Dm + i] = (_Float16)wv[i];
    w16[3 * Dm * Dm + i] = (_Float16)wo[i];
  }
}

// ---- Kernel 1: fused QKV projection (x @ W.T + b), head-split layouts ----
// q16,k16: [B,H,S,HD] f16 (q pre-scaled by 1/sqrt(HD)); vt16: [B,H,HD,S] f16
__global__ __launch_bounds__(256) void qkv_proj_kernel(
    const float* __restrict__ query, const float* __restrict__ key_, const float* __restrict__ value,
    const _Float16* __restrict__ w16,
    const float* __restrict__ bq, const float* __restrict__ bk, const float* __restrict__ bv,
    _Float16* __restrict__ q16, _Float16* __restrict__ k16, _Float16* __restrict__ vt16) {
  __shared__ _Float16 xtile[16 * 264];   // f16 A tile, converted once per block
  const int which = blockIdx.y;          // 0=q 1=k 2=v
  const int row0 = blockIdx.x * 16;      // flattened B*S row tile
  const int wave = threadIdx.x >> 5;
  const int lane = threadIdx.x & 31;
  const int n0 = wave * 32;              // 8 waves cover 256 output cols

  const float* x    = (which == 0) ? query : (which == 1) ? key_ : value;
  const float* bias = (which == 0) ? bq : (which == 1) ? bk : bv;
  const _Float16* W = w16 + (size_t)which * Dm * Dm;   // row j = output col j

  stage_a_tile_f16(xtile, x + (size_t)row0 * Dm, Dm);
  __syncthreads();

  v8f c0 = {}, c1 = {};
#pragma unroll
  for (int kb = 0; kb < Dm; kb += 32) {
    v16h a  = load_a_h(xtile, 264, kb);
    v16h b0 = load_b_h(W, Dm, n0, kb);
    v16h b1 = load_b_h(W, Dm, n0 + 16, kb);
    c0 = wmma16(a, b0, c0);
    c1 = wmma16(a, b1, c1);
  }

  const int nn = lane & 15, half = lane >> 4;
#pragma unroll
  for (int t = 0; t < 2; ++t) {
    v8f c = (t == 0) ? c0 : c1;
    const int col = n0 + t * 16 + nn;
    const int h = col / HD, hd = col % HD;
#pragma unroll
    for (int v = 0; v < 8; ++v) {
      const int r = row0 + v + 8 * half;
      const int b = r / Sn, s = r % Sn;
      float val = c[v] + bias[col];
      if (which == 0)
        q16[(((size_t)b * Hn + h) * Sn + s) * HD + hd] = (_Float16)(val * 0.125f); // 1/sqrt(64)
      else if (which == 1)
        k16[(((size_t)b * Hn + h) * Sn + s) * HD + hd] = (_Float16)val;
      else
        vt16[(((size_t)b * Hn + h) * HD + hd) * Sn + s] = (_Float16)val;
    }
  }
}

// ---- Kernel 2: flash attention, async double-buffered K/V in LDS ---------
#define PRS 40   // padded LDS row stride for 16x32 P tile
#define KLD 72   // padded row stride, K tile 32x64   (lane stride 36 banks mod 64)
#define VLD 40   // padded row stride, Vt tile 64x32  (lane stride 20 banks mod 64)
__global__ __launch_bounds__(256) void flash_attn_kernel(
    const _Float16* __restrict__ q16, const _Float16* __restrict__ k16,
    const _Float16* __restrict__ vt16, float* __restrict__ attn) {
  __shared__ _Float16 kbuf[2][32 * KLD];
  __shared__ _Float16 vbuf[2][64 * VLD];
  __shared__ _Float16 plds[8 * 16 * PRS];  // per-wave 16x32 P tile

  const int wave = threadIdx.x >> 5;
  const int lane = threadIdx.x & 31;
  const int nn = lane & 15, half = lane >> 4;
  const int bh = blockIdx.y;
  const int b = bh >> 2, h = bh & 3;                  // H = 4
  const int q0 = blockIdx.x * 128 + wave * 16;        // each wave: 16 query rows

  const _Float16* Q  = q16 + ((size_t)bh * Sn + q0) * HD;
  const _Float16* K  = k16 + (size_t)bh * Sn * HD;
  const _Float16* Vt = vt16 + (size_t)bh * HD * Sn;
  _Float16* pbase = plds + wave * 16 * PRS;

  // cooperative async staging of one 32-key chunk of K and Vt (1 b128/thread each)
  auto stage = [&](int buf, int jn) {
    const int t = threadIdx.x;
    {  // K chunk: 32 rows x 64 halfs
      const int r = t >> 3, sg = t & 7;
      async_b128(&kbuf[buf][r * KLD + sg * 8], K + (size_t)(jn + r) * HD + sg * 8);
    }
    {  // Vt chunk: 64 rows x 32 halfs
      const int d = t >> 2, sg = t & 3;
      async_b128(&vbuf[buf][d * VLD + sg * 8], Vt + (size_t)d * Sn + jn + sg * 8);
    }
  };

  const v16h qa0 = load_a_h(Q, HD, 0);    // d = 0..31
  const v16h qa1 = load_a_h(Q, HD, 32);   // d = 32..63

  float Mrow[8], Lrow[8];
  v8f o0 = {}, o1 = {}, o2 = {}, o3 = {};
#pragma unroll
  for (int v = 0; v < 8; ++v) { Mrow[v] = -1e30f; Lrow[v] = 0.f; }

  stage(0, 0);
  wait_async0();
  __syncthreads();

  for (int j = 0; j < Sn; j += 32) {
    const int buf = (j >> 5) & 1;
    if (j + 32 < Sn) stage(buf ^ 1, j + 32);   // prefetch next chunk (async)

    // scores: S = (Q/sqrt(d)) @ K^T  -> two 16x16 f32 fragments (from LDS)
    v8f s0 = {}, s1 = {};
    {
      v16h kb;
      kb = load_b_h(kbuf[buf], KLD, 0, 0);   s0 = wmma16(qa0, kb, s0);
      kb = load_b_h(kbuf[buf], KLD, 0, 32);  s0 = wmma16(qa1, kb, s0);
      kb = load_b_h(kbuf[buf], KLD, 16, 0);  s1 = wmma16(qa0, kb, s1);
      kb = load_b_h(kbuf[buf], KLD, 16, 32); s1 = wmma16(qa1, kb, s1);
    }

    // online softmax: row reductions across the 16 lanes of each half-wave
    float p0[8], p1[8];
#pragma unroll
    for (int v = 0; v < 8; ++v) {
      float tm = fmaxf(s0[v], s1[v]);
#pragma unroll
      for (int m = 1; m <= 8; m <<= 1) tm = fmaxf(tm, __shfl_xor(tm, m, 32));
      const float nM = fmaxf(Mrow[v], tm);
      const float alpha = __expf(Mrow[v] - nM);
      Mrow[v] = nM;
      p0[v] = __expf(s0[v] - nM);
      p1[v] = __expf(s1[v] - nM);
      float rs = p0[v] + p1[v];
#pragma unroll
      for (int m = 1; m <= 8; m <<= 1) rs += __shfl_xor(rs, m, 32);
      Lrow[v] = Lrow[v] * alpha + rs;
      o0[v] *= alpha; o1[v] *= alpha; o2[v] *= alpha; o3[v] *= alpha;
    }

    // restage P: C-layout -> LDS row-major -> A-layout fragment.
    // plds is per-wave; same-wave DS ops execute in order, so no barrier needed.
#pragma unroll
    for (int v = 0; v < 8; ++v) {
      const int r = v + 8 * half;
      pbase[r * PRS + nn]      = (_Float16)p0[v];
      pbase[r * PRS + 16 + nn] = (_Float16)p1[v];
    }
    compiler_fence();

    const v16h pa = load_a_h(pbase, PRS, 0);
    o0 = wmma16(pa, load_b_h(vbuf[buf], VLD, 0,  0), o0);
    o1 = wmma16(pa, load_b_h(vbuf[buf], VLD, 16, 0), o1);
    o2 = wmma16(pa, load_b_h(vbuf[buf], VLD, 32, 0), o2);
    o3 = wmma16(pa, load_b_h(vbuf[buf], VLD, 48, 0), o3);

    // next buffer staged + everyone done reading current buffer
    wait_async0();
    __syncthreads();
  }

  // epilogue: normalize by row sum, scatter heads back to [B,S,D] fp32
#pragma unroll
  for (int v = 0; v < 8; ++v) {
    const float inv = 1.f / Lrow[v];
    const int s = q0 + v + 8 * half;
    float* op = attn + ((size_t)b * Sn + s) * Dm + h * HD + nn;
    op[0]  = o0[v] * inv;
    op[16] = o1[v] * inv;
    op[32] = o2[v] * inv;
    op[48] = o3[v] * inv;
  }
}

// ---- Kernel 3: output projection + fused LayerNorm -----------------------
__global__ __launch_bounds__(256) void out_proj_ln_kernel(
    const float* __restrict__ attn, const _Float16* __restrict__ wo16,
    const float* __restrict__ bo, const float* __restrict__ gamma,
    const float* __restrict__ beta, float* __restrict__ out) {
  __shared__ _Float16 atile[16 * 264];  // f16 A tile (converted once)
  __shared__ float    rtile[16 * 264];  // fp32 result tile for LN
  const int row0 = blockIdx.x * 16;
  const int wave = threadIdx.x >> 5;
  const int lane = threadIdx.x & 31;
  const int nn = lane & 15, half = lane >> 4;
  const int n0 = wave * 32;

  stage_a_tile_f16(atile, attn + (size_t)row0 * Dm, Dm);
  __syncthreads();

  v8f c0 = {}, c1 = {};
#pragma unroll
  for (int kb = 0; kb < Dm; kb += 32) {
    v16h a = load_a_h(atile, 264, kb);
    c0 = wmma16(a, load_b_h(wo16, Dm, n0, kb), c0);
    c1 = wmma16(a, load_b_h(wo16, Dm, n0 + 16, kb), c1);
  }

#pragma unroll
  for (int v = 0; v < 8; ++v) {
    const int r = v + 8 * half;
    rtile[r * 264 + n0 + nn]      = c0[v] + bo[n0 + nn];
    rtile[r * 264 + n0 + 16 + nn] = c1[v] + bo[n0 + 16 + nn];
  }
  __syncthreads();

  // LayerNorm: 16 threads per row, butterfly within 16-lane groups
  const int row = threadIdx.x >> 4;
  const int cbase = threadIdx.x & 15;
  float sum = 0.f, sq = 0.f;
#pragma unroll
  for (int i = 0; i < 16; ++i) {
    const float xv = rtile[row * 264 + cbase + i * 16];
    sum += xv; sq += xv * xv;
  }
#pragma unroll
  for (int m = 1; m <= 8; m <<= 1) {
    sum += __shfl_xor(sum, m, 32);
    sq  += __shfl_xor(sq, m, 32);
  }
  const float mean = sum * (1.f / 256.f);
  const float var  = sq * (1.f / 256.f) - mean * mean;
  const float rstd = rsqrtf(var + 1e-5f);
#pragma unroll
  for (int i = 0; i < 16; ++i) {
    const int col = cbase + i * 16;
    const float xv = rtile[row * 264 + col];
    out[(size_t)(row0 + row) * Dm + col] = (xv - mean) * rstd * gamma[col] + beta[col];
  }
}

// ---- launcher ------------------------------------------------------------
extern "C" void kernel_launch(void* const* d_in, const int* in_sizes, int n_in,
                              void* d_out, int out_size, void* d_ws, size_t ws_size,
                              hipStream_t stream) {
  (void)in_sizes; (void)n_in; (void)out_size; (void)ws_size;
  const float* query = (const float*)d_in[0];
  const float* key_  = (const float*)d_in[1];
  const float* value = (const float*)d_in[2];
  const float* wq = (const float*)d_in[3];
  const float* bq = (const float*)d_in[4];
  const float* wk = (const float*)d_in[5];
  const float* bk = (const float*)d_in[6];
  const float* wv = (const float*)d_in[7];
  const float* bv = (const float*)d_in[8];
  const float* wo = (const float*)d_in[9];
  const float* bo = (const float*)d_in[10];
  const float* gamma = (const float*)d_in[11];
  const float* beta  = (const float*)d_in[12];
  float* out = (float*)d_out;

  char* ws = (char*)d_ws;
  _Float16* w16  = (_Float16*)ws;  ws += (size_t)4 * Dm * Dm * sizeof(_Float16);
  _Float16* q16  = (_Float16*)ws;  ws += (size_t)BS * Dm * sizeof(_Float16);
  _Float16* k16  = (_Float16*)ws;  ws += (size_t)BS * Dm * sizeof(_Float16);
  _Float16* vt16 = (_Float16*)ws;  ws += (size_t)BS * Dm * sizeof(_Float16);
  float*    attn = (float*)ws;     ws += (size_t)BS * Dm * sizeof(float);

  cvt_weights_kernel<<<(Dm * Dm + 255) / 256, 256, 0, stream>>>(wq, wk, wv, wo, w16);
  qkv_proj_kernel<<<dim3(BS / 16, 3), 256, 0, stream>>>(query, key_, value, w16,
                                                        bq, bk, bv, q16, k16, vt16);
  flash_attn_kernel<<<dim3(Sn / 128, Bn * Hn), 256, 0, stream>>>(q16, k16, vt16, attn);
  out_proj_ln_kernel<<<BS / 16, 256, 0, stream>>>(attn, w16 + 3 * Dm * Dm, bo, gamma, beta, out);
}
